// SACRSN_v42_23536420782616
// MI455X (gfx1250) — matrix operand order repaired and checked by reading, not compiled, not run on hardware
//
#include <hip/hip_runtime.h>
#include <stdint.h>

// Problem constants (match reference)
constexpr int Bb = 128, Tt = 32, Dd = 64, Vv = 4096, Ss = 32, Rr = 8, K2 = 128;
constexpr float PONDER = 0.01f;
constexpr float TWO_PI_F = 6.28318530717958647692f;

typedef __attribute__((ext_vector_type(16))) __bf16 v16bf;
typedef __attribute__((ext_vector_type(8)))  float  v8f;

union BF16Frag { unsigned short s[16]; unsigned u32[8]; uint4 q[2]; v16bf v; };
union AccU     { v8f v; float f[8]; };

__device__ __forceinline__ unsigned short f2bf(float f) {
  unsigned u = __float_as_uint(f);
  unsigned r = u + 0x7FFFu + ((u >> 16) & 1u);  // round-to-nearest-even
  return (unsigned short)(r >> 16);
}

// 16-bit A-matrix 16x32 layout (ISA 7.12.2): lane<16 -> K {0..7,16..23}, lane>=16 -> +8
__device__ __forceinline__ int a_k(int ks, int hlf, int e) {
  int dw = e >> 1, odd = e & 1;
  return (dw < 4) ? (ks * 32 + hlf * 8 + dw * 2 + odd)
                  : (ks * 32 + 16 + hlf * 8 + (dw - 4) * 2 + odd);
}
// 16-bit B-matrix 32x16 layout: same K order as A (lane<16 -> K {0..7,16..23}, lane>=16 -> +8)
__device__ __forceinline__ int b_k(int ks, int hlf, int e) {
  return a_k(ks, hlf, e);
}

// ---------------------------------------------------------------------------
// Kernel 0: precompute vq_sq, and bf16 fragment-major layouts of vq_emb^T and dec_w
// ---------------------------------------------------------------------------
__global__ void __launch_bounds__(256) k_prep(const float* __restrict__ vq_emb,
                                              const float* __restrict__ dec_w,
                                              unsigned* __restrict__ vq_frag,
                                              unsigned* __restrict__ dec_frag,
                                              float* __restrict__ vq_sq) {
  const int NF = (Vv / 16) * 4 * 32;  // tiles * ksteps * lanes = 32768
  int gid = blockIdx.x * blockDim.x + threadIdx.x;
  if (gid < Vv) {
    float s = 0.f;
    for (int k = 0; k < K2; ++k) { float x = vq_emb[(size_t)gid * K2 + k]; s += x * x; }
    vq_sq[gid] = s;
  }
  int f0 = gid - Vv;
  if (f0 >= 0 && f0 < NF) {                     // B-operand = vq_emb^T : B[k][n] = vq_emb[n][k]
    int lane = f0 & 31, tmp = f0 >> 5;
    int ks = tmp & 3, tile = tmp >> 2;
    int hlf = lane >> 4, col = tile * 16 + (lane & 15);
    unsigned* dst = vq_frag + (size_t)f0 * 8;
    for (int dw = 0; dw < 8; ++dw) {
      int k0 = b_k(ks, hlf, 2 * dw), k1 = b_k(ks, hlf, 2 * dw + 1);
      dst[dw] = (unsigned)f2bf(vq_emb[(size_t)col * K2 + k0]) |
                ((unsigned)f2bf(vq_emb[(size_t)col * K2 + k1]) << 16);
    }
  }
  int f1 = gid - Vv - NF;
  if (f1 >= 0 && f1 < NF) {                     // B-operand = dec_w : B[k][n] = dec_w[k][n]
    int lane = f1 & 31, tmp = f1 >> 5;
    int ks = tmp & 3, tile = tmp >> 2;
    int hlf = lane >> 4, col = tile * 16 + (lane & 15);
    unsigned* dst = dec_frag + (size_t)f1 * 8;
    for (int dw = 0; dw < 8; ++dw) {
      int k0 = b_k(ks, hlf, 2 * dw), k1 = b_k(ks, hlf, 2 * dw + 1);
      dst[dw] = (unsigned)f2bf(dec_w[(size_t)k0 * Vv + col]) |
                ((unsigned)f2bf(dec_w[(size_t)k1 * Vv + col]) << 16);
    }
  }
}

// ---------------------------------------------------------------------------
// Kernel 1: full T x R recurrence. 8 workgroups x 16 batch rows, 8 waves each.
// ---------------------------------------------------------------------------
__global__ void __launch_bounds__(256) k_main(
    const int* __restrict__ x_seq, const float* __restrict__ enc_emb,
    const float* __restrict__ vq_emb,
    const float* __restrict__ nrg, const float* __restrict__ nrb,
    const float* __restrict__ nig, const float* __restrict__ nib,
    const float* __restrict__ proj_w, const float* __restrict__ proj_b,
    const float* __restrict__ mg_w, const float* __restrict__ mg_b,
    const float* __restrict__ arb_w, const float* __restrict__ arb_b,
    const float* __restrict__ halt_bias_p, const float* __restrict__ input_gate_p,
    const unsigned* __restrict__ vq_frag, const float* __restrict__ vq_sq,
    float* __restrict__ ws_flat, int* __restrict__ idx_ws, int* __restrict__ mask_ws,
    float* __restrict__ accvq_ws, float* __restrict__ accpond_ws, float* __restrict__ accph_ws,
    float* __restrict__ mem_r_out, float* __restrict__ mem_i_out) {
  __shared__ float flat[16][K2];
  __shared__ float gwr[16][Dd], gwi[16][Dd];
  __shared__ float pang[16][Dd];
  __shared__ float wrs[16][Dd], wis[16][Dd];
  __shared__ float mrs[16][Dd], mis[16][Dd];
  __shared__ float attn[16][Ss];
  __shared__ float redv[128];
  __shared__ int   redi[128];
  __shared__ float gp[256], dfp[256];
  __shared__ float rowsq[16], vqls[16], activef[16], gatev[16], wgv[16];
  __shared__ float gsm[16][3], glog[16][3];
  __shared__ int   idxrow[16];
  __shared__ float accvq_s[16], accpond_s[16], accph_s[16];

  const int tid = threadIdx.x, lane = tid & 31, wave = tid >> 5;
  const int wg = blockIdx.x;
  const float igv = input_gate_p[0];
  const float alpha = 1.f / (1.f + expf(-igv));
  const float hb = halt_bias_p[0];
  const float bias = (hb > 20.f) ? hb : log1pf(expf(hb));  // softplus

  // zero the persistent memory (lives directly in d_out; offset returns to 0 after T=32 rolls)
  for (int i = tid; i < 16 * Ss * Dd; i += 256) {
    int row = i / (Ss * Dd), rem = i % (Ss * Dd);
    mem_r_out[((size_t)(wg * 16 + row)) * Ss * Dd + rem] = 0.f;
    mem_i_out[((size_t)(wg * 16 + row)) * Ss * Dd + rem] = 0.f;
  }
  for (int i = tid; i < 16 * Dd; i += 256) { gwr[i / Dd][i % Dd] = 0.f; gwi[i / Dd][i % Dd] = 0.f; }
  __threadfence_block();
  __syncthreads();

  int memoff = 0;
  for (int t = 0; t < Tt; ++t) {
    // input blend + prev angle
    for (int i = tid; i < 16 * Dd; i += 256) {
      int row = i / Dd, d = i % Dd, b = wg * 16 + row;
      int tok = x_seq[b * Tt + t];
      float er = enc_emb[(size_t)tok * K2 + d];
      float ei = enc_emb[(size_t)tok * K2 + Dd + d];
      float gr = alpha * gwr[row][d] + (1.f - alpha) * er;
      float gi = alpha * gwi[row][d] + (1.f - alpha) * ei;
      gwr[row][d] = gr; gwi[row][d] = gi;
      pang[row][d] = atan2f(gi, gr);
    }
    if (tid < 16) { activef[tid] = 1.f; accvq_s[tid] = 0.f; accpond_s[tid] = 0.f; accph_s[tid] = 0.f; }
    __syncthreads();

    for (int it = 0; it < Rr; ++it) {
      // ---- 1) LayerNorm (wave handles rows 2w, 2w+1; 2 elems/lane) ----
      for (int rr = 0; rr < 2; ++rr) {
        int row = wave * 2 + rr;
        float xr0 = gwr[row][lane], xr1 = gwr[row][lane + 32];
        float xi0 = gwi[row][lane], xi1 = gwi[row][lane + 32];
        float sr = xr0 + xr1, si = xi0 + xi1;
        for (int off = 16; off >= 1; off >>= 1) { sr += __shfl_xor(sr, off, 32); si += __shfl_xor(si, off, 32); }
        float mur = sr * (1.f / 64.f), mui = si * (1.f / 64.f);
        float dr0 = xr0 - mur, dr1 = xr1 - mur, di0 = xi0 - mui, di1 = xi1 - mui;
        float vr = dr0 * dr0 + dr1 * dr1, vi = di0 * di0 + di1 * di1;
        for (int off = 16; off >= 1; off >>= 1) { vr += __shfl_xor(vr, off, 32); vi += __shfl_xor(vi, off, 32); }
        float isr = rsqrtf(vr * (1.f / 64.f) + 1e-5f), isi = rsqrtf(vi * (1.f / 64.f) + 1e-5f);
        float cr0 = dr0 * isr * nrg[lane]      + nrb[lane];
        float cr1 = dr1 * isr * nrg[lane + 32] + nrb[lane + 32];
        float ci0 = di0 * isi * nig[lane]      + nib[lane];
        float ci1 = di1 * isi * nig[lane + 32] + nib[lane + 32];
        flat[row][lane] = cr0;      flat[row][lane + 32] = cr1;
        flat[row][Dd + lane] = ci0; flat[row][Dd + lane + 32] = ci1;
        float sq = cr0 * cr0 + cr1 * cr1 + ci0 * ci0 + ci1 * ci1;
        for (int off = 16; off >= 1; off >>= 1) sq += __shfl_xor(sq, off, 32);
        if (lane == 0) rowsq[row] = sq;
      }
      __syncthreads();

      // ---- 2) VQ distances via WMMA bf16 (flat @ vq_emb^T), argmin ----
      const int hlf = lane >> 4;
      BF16Frag afr[4];
      for (int ks = 0; ks < 4; ++ks) {
        int m = lane & 15;
        for (int dw = 0; dw < 8; ++dw) {
          int k0 = a_k(ks, hlf, 2 * dw), k1 = a_k(ks, hlf, 2 * dw + 1);
          afr[ks].u32[dw] = (unsigned)f2bf(flat[m][k0]) | ((unsigned)f2bf(flat[m][k1]) << 16);
        }
      }
      float bestv[8]; int besti[8];
      for (int j = 0; j < 8; ++j) { bestv[j] = 3.4e38f; besti[j] = 0x7fffffff; }
      for (int nt = 0; nt < 32; ++nt) {
        int tile = wave * 32 + nt;
        AccU acc; for (int j = 0; j < 8; ++j) acc.f[j] = 0.f;
        if (nt < 31)
          __builtin_prefetch((const void*)(vq_frag + (((size_t)(tile + 1) * 4) * 32 + lane) * 8), 0, 1);
        for (int ks = 0; ks < 4; ++ks) {
          BF16Frag bfr;
          const uint4* p = (const uint4*)(vq_frag + (((size_t)tile * 4 + ks) * 32 + lane) * 8);
          bfr.q[0] = p[0]; bfr.q[1] = p[1];
          acc.v = __builtin_amdgcn_wmma_f32_16x16x32_bf16(false, afr[ks].v, false, bfr.v,
                                                          (short)0, acc.v, false, false);
        }
        int col = tile * 16 + (lane & 15);
        float csq = vq_sq[col];
        for (int j = 0; j < 8; ++j) {
          float d2p = csq - 2.f * acc.f[j];   // rowsq added later (row-constant)
          if (d2p < bestv[j] || (d2p == bestv[j] && col < besti[j])) { bestv[j] = d2p; besti[j] = col; }
        }
      }
      for (int j = 0; j < 8; ++j)
        for (int off = 8; off >= 1; off >>= 1) {
          float ov = __shfl_xor(bestv[j], off, 32);
          int oi = __shfl_xor(besti[j], off, 32);
          if (ov < bestv[j] || (ov == bestv[j] && oi < besti[j])) { bestv[j] = ov; besti[j] = oi; }
        }
      if ((lane & 15) == 0)
        for (int j = 0; j < 8; ++j) { redv[wave * 16 + hlf * 8 + j] = bestv[j]; redi[wave * 16 + hlf * 8 + j] = besti[j]; }
      __syncthreads();
      if (tid < 16) {
        float bv = redv[tid]; int bi = redi[tid];
        for (int w = 1; w < 8; ++w) {
          float ov = redv[w * 16 + tid]; int oi = redi[w * 16 + tid];
          if (ov < bv || (ov == bv && oi < bi)) { bv = ov; bi = oi; }
        }
        float d2 = rowsq[tid] + bv;
        vqls[tid] = d2 * (1.25f / (float)K2);
        idxrow[tid] = bi;
        int b = wg * 16 + tid;
        idx_ws[(t * Rr + it) * Bb + b]  = bi;
        mask_ws[(t * Rr + it) * Bb + b] = activef[tid] > 0.5f ? 1 : 0;
      }
      __syncthreads();

      // ---- 3) complex projections q,k,w + gate; arbiter logits ----
      {
        int row = tid >> 4, og = tid & 15;
        float gpart = 0.f;
        for (int jo = 0; jo < 4; ++jo) {
          int o = og * 4 + jo;
          float qr = 0.f, qi = 0.f, kr = 0.f, ki = 0.f;
          for (int p = 0; p < 3; ++p) {
            const float* Wr = proj_w + ((size_t)(p * 2 + 0) * Dd) * Dd;
            const float* Wi = proj_w + ((size_t)(p * 2 + 1) * Dd) * Dd;
            float br = proj_b[(p * 2 + 0) * Dd + o];
            float bi = proj_b[(p * 2 + 1) * Dd + o];
            float s_rr = 0.f, s_ii = 0.f, s_ir = 0.f, s_ri = 0.f;
            for (int d = 0; d < Dd; ++d) {
              float cr = flat[row][d], ci = flat[row][Dd + d];
              float wrv = Wr[d * Dd + o], wiv = Wi[d * Dd + o];
              s_rr += cr * wrv; s_ii += ci * wiv; s_ir += ci * wrv; s_ri += cr * wiv;
            }
            float re = s_rr + br - (s_ii + bi);
            float im = s_ir + br + (s_ri + bi);
            if (p == 0)      { qr = re; qi = im; }
            else if (p == 1) { kr = re; ki = im; }
            else             { wrs[row][o] = re; wis[row][o] = im; }
          }
          gpart += qr * kr + qi * ki;
        }
        gp[tid] = gpart;
      }
      if (tid < 48) {
        int row = tid / 3, c = tid % 3;
        float s = arb_b[c];
        for (int k = 0; k < K2; ++k) s += flat[row][k] * arb_w[k * 3 + c];
        glog[row][c] = s;
      }
      __syncthreads();
      if (tid < 16) {
        float s = 0.f; for (int i = 0; i < 16; ++i) s += gp[tid * 16 + i];
        gatev[tid] = 1.f / (1.f + expf(-s));
        float l0 = glog[tid][0], l1 = glog[tid][1], l2 = glog[tid][2];
        float mx = fmaxf(l0, fmaxf(l1, l2));
        float e0 = expf(l0 - mx), e1 = expf(l1 - mx), e2 = expf(l2 - mx);
        float inv = 1.f / (e0 + e1 + e2);
        gsm[tid][0] = e0 * inv; gsm[tid][1] = e1 * inv; gsm[tid][2] = e2 * inv;
      }

      // ---- 4) memory attention (lane = slot) ----
      for (int rr = 0; rr < 2; ++rr) {
        int row = wave * 2 + rr, b = wg * 16 + row;
        int sp = (lane + memoff) & (Ss - 1);
        const float* mr = mem_r_out + ((size_t)b * Ss + sp) * Dd;
        const float* mi = mem_i_out + ((size_t)b * Ss + sp) * Dd;
        float sim = 0.f;
        for (int d = 0; d < Dd; ++d) sim += mr[d] * flat[row][d] + mi[d] * flat[row][Dd + d];
        float mx = sim;
        for (int off = 16; off >= 1; off >>= 1) mx = fmaxf(mx, __shfl_xor(mx, off, 32));
        float e = expf(sim - mx);
        float ssum = e;
        for (int off = 16; off >= 1; off >>= 1) ssum += __shfl_xor(ssum, off, 32);
        attn[row][lane] = e / ssum;
      }
      __syncthreads();
      for (int i = tid; i < 16 * Dd; i += 256) {
        int row = i / Dd, d = i % Dd, b = wg * 16 + row;
        float smr = 0.f, smi = 0.f;
        for (int s = 0; s < Ss; ++s) {
          int sp = (s + memoff) & (Ss - 1);
          float a = attn[row][s];
          smr += a * mem_r_out[((size_t)b * Ss + sp) * Dd + d];
          smi += a * mem_i_out[((size_t)b * Ss + sp) * Dd + d];
        }
        mrs[row][d] = smr; mis[row][d] = smi;
      }
      __syncthreads();

      // ---- 5) arbitrate, update, phase, halting ----
      {
        int row = tid >> 4, dg = tid & 15;
        float dsum = 0.f;
        float g0 = gsm[row][0], g1 = gsm[row][1], g2 = gsm[row][2];
        float gt = gatev[row];
        const float* zq = vq_emb + (size_t)idxrow[row] * K2;
        bool act = activef[row] > 0.5f;
        for (int j = 0; j < 4; ++j) {
          int d = dg * 4 + j;
          float cr = flat[row][d], ci = flat[row][Dd + d];
          float ur = g0 * (wrs[row][d] * gt) + g1 * mrs[row][d] + g2 * zq[d];
          float ui = g0 * (wis[row][d] * gt) + g1 * mis[row][d] + g2 * zq[Dd + d];
          float c_r = 0.6f * cr + 0.4f * ur;
          float c_i = 0.6f * ci + 0.4f * ui;
          float ang = atan2f(c_i, c_r);
          float df = fabsf(ang - pang[row][d]);
          df = fminf(df, TWO_PI_F - df);
          dsum += df;
          pang[row][d] = ang;
          if (act) { gwr[row][d] = c_r; gwi[row][d] = c_i; }
        }
        dfp[tid] = dsum;
      }
      __syncthreads();
      if (tid < 16) {
        float s = 0.f; for (int i = 0; i < 16; ++i) s += dfp[tid * 16 + i];
        float dmean = s * (1.f / 64.f);
        float a = activef[tid];
        accph_s[tid]   += a * dmean;
        accpond_s[tid] += a * PONDER;
        if (a > 0.5f) accvq_s[tid] = vqls[tid];
        float halt = 1.f / (1.f + expf(-(bias - vqls[tid])));
        activef[tid] = a * (halt > 0.5f ? 0.f : 1.f);
      }
      __syncthreads();
    }  // R

    // per-timestep outputs + memory roll/write (circular offset)
    for (int i = tid; i < 16 * K2; i += 256) {
      int row = i / K2, k = i % K2, b = wg * 16 + row;
      float v = (k < Dd) ? gwr[row][k] : gwi[row][k - Dd];
      ws_flat[((size_t)b * Tt + t) * K2 + k] = v;
    }
    if (tid < 16) {
      int b = wg * 16 + tid;
      accvq_ws[t * Bb + b]   = accvq_s[tid];
      accpond_ws[t * Bb + b] = accpond_s[tid];
      accph_ws[t * Bb + b]   = accph_s[tid];
      float s = mg_b[0];
      for (int k = 0; k < Dd; ++k) s += gwr[tid][k] * mg_w[k];
      for (int k = 0; k < Dd; ++k) s += gwi[tid][k] * mg_w[Dd + k];
      wgv[tid] = 1.f / (1.f + expf(-s));
    }
    __syncthreads();
    memoff = (memoff + Ss - 1) & (Ss - 1);
    for (int i = tid; i < 16 * Dd; i += 256) {
      int row = i / Dd, d = i % Dd, b = wg * 16 + row;
      float w = wgv[row];
      size_t p0 = ((size_t)b * Ss + memoff) * Dd + d;
      mem_r_out[p0] = w * gwr[row][d] + (1.f - w) * mem_r_out[p0];
      mem_i_out[p0] = w * gwi[row][d] + (1.f - w) * mem_i_out[p0];
    }
    __threadfence_block();
    __syncthreads();
  }  // T
}

// ---------------------------------------------------------------------------
// Kernel 2: decode logits = flat_out @ dec_w + dec_b via WMMA bf16
// ---------------------------------------------------------------------------
__global__ void __launch_bounds__(256) k_decode(const float* __restrict__ ws_flat,
                                                const unsigned* __restrict__ dec_frag,
                                                const float* __restrict__ dec_b,
                                                float* __restrict__ logits) {
  __shared__ float rows[16][K2];
  const int tid = threadIdx.x, lane = tid & 31, wave = tid >> 5;
  const int mtile = blockIdx.x;
  for (int i = tid; i < 16 * K2; i += 256)
    rows[i / K2][i % K2] = ws_flat[(size_t)mtile * 16 * K2 + i];
  __syncthreads();
  const int hlf = lane >> 4, m = lane & 15;
  BF16Frag afr[4];
  for (int ks = 0; ks < 4; ++ks)
    for (int dw = 0; dw < 8; ++dw) {
      int k0 = a_k(ks, hlf, 2 * dw), k1 = a_k(ks, hlf, 2 * dw + 1);
      afr[ks].u32[dw] = (unsigned)f2bf(rows[m][k0]) | ((unsigned)f2bf(rows[m][k1]) << 16);
    }
  for (int nt = 0; nt < 32; ++nt) {
    int tile = wave * 32 + nt;
    AccU acc; for (int j = 0; j < 8; ++j) acc.f[j] = 0.f;
    if (nt < 31)
      __builtin_prefetch((const void*)(dec_frag + (((size_t)(tile + 1) * 4) * 32 + lane) * 8), 0, 1);
    for (int ks = 0; ks < 4; ++ks) {
      BF16Frag bfr;
      const uint4* p = (const uint4*)(dec_frag + (((size_t)tile * 4 + ks) * 32 + lane) * 8);
      bfr.q[0] = p[0]; bfr.q[1] = p[1];
      acc.v = __builtin_amdgcn_wmma_f32_16x16x32_bf16(false, afr[ks].v, false, bfr.v,
                                                      (short)0, acc.v, false, false);
    }
    int col = tile * 16 + m;
    float bv = dec_b[col];
    for (int j = 0; j < 8; ++j) {
      int rowi = mtile * 16 + hlf * 8 + j;
      logits[(size_t)rowi * Vv + col] = acc.f[j] + bv;
    }
  }
}

// ---------------------------------------------------------------------------
// Kernel 3: batch-entropy stats from recorded idx/active masks
// ---------------------------------------------------------------------------
__global__ void __launch_bounds__(256) k_stats(const int* __restrict__ idx_ws,
                                               const int* __restrict__ mask_ws,
                                               const float* __restrict__ accvq_ws,
                                               const float* __restrict__ accpond_ws,
                                               const float* __restrict__ accph_ws,
                                               float* __restrict__ out_stats) {
  __shared__ int   idx_s[Rr][Bb];
  __shared__ int   msk_s[Rr][Bb];
  __shared__ float ent_s[Rr], ppx_s[Rr];
  __shared__ float entb[Bb], ppxb[Bb];
  __shared__ float acc5[5];
  const int tid = threadIdx.x, lane = tid & 31;
  if (tid < 5) acc5[tid] = 0.f;
  for (int tt = 0; tt < Tt; ++tt) {
    for (int i = tid; i < Rr * Bb; i += 256) {
      idx_s[i / Bb][i % Bb] = idx_ws[tt * Rr * Bb + i];
      msk_s[i / Bb][i % Bb] = mask_ws[tt * Rr * Bb + i];
    }
    __syncthreads();
    {
      int r = tid >> 5;  // 8 groups of 32 lanes, one r each
      float part = 0.f;
      for (int b = lane; b < Bb; b += 32) {
        int v = idx_s[r][b];
        int c = 0;
        for (int b2 = 0; b2 < Bb; ++b2) c += (idx_s[r][b2] == v) ? 1 : 0;
        part += logf((float)c * (1.f / (float)Bb) + 1e-10f);
      }
      for (int off = 16; off >= 1; off >>= 1) part += __shfl_xor(part, off, 32);
      if (lane == 0) { float e = -part * (1.f / (float)Bb); ent_s[r] = e; ppx_s[r] = expf(e); }
    }
    __syncthreads();
    if (tid < Bb) {
      float ae = 0.f, ap = 0.f;
      for (int r = 0; r < Rr; ++r)
        if (msk_s[r][tid]) { ae = ent_s[r]; ap = ppx_s[r]; }
      entb[tid] = ae; ppxb[tid] = ap;
    }
    __syncthreads();
    if (tid == 0) {
      float s0 = 0.f, s1 = 0.f, s2 = 0.f, s3 = 0.f, s4 = 0.f;
      for (int b = 0; b < Bb; ++b) {
        s0 += accvq_ws[tt * Bb + b];
        s1 += ppxb[b];
        s2 += entb[b];
        s3 += accpond_ws[tt * Bb + b];
        s4 += accph_ws[tt * Bb + b];
      }
      acc5[0] += s0; acc5[1] += s1; acc5[2] += s2; acc5[3] += s3; acc5[4] += s4;
    }
    __syncthreads();
  }
  if (tid < 5) out_stats[tid] = acc5[tid] * (1.f / (float)(Bb * Tt));
}

// ---------------------------------------------------------------------------
extern "C" void kernel_launch(void* const* d_in, const int* in_sizes, int n_in,
                              void* d_out, int out_size, void* d_ws, size_t ws_size,
                              hipStream_t stream) {
  (void)in_sizes; (void)n_in; (void)out_size; (void)ws_size;
  const int*   x_seq      = (const int*)d_in[0];
  const float* enc_emb    = (const float*)d_in[1];
  const float* vq_emb     = (const float*)d_in[2];
  const float* nrg        = (const float*)d_in[3];
  const float* nrb        = (const float*)d_in[4];
  const float* nig        = (const float*)d_in[5];
  const float* nib        = (const float*)d_in[6];
  const float* proj_w     = (const float*)d_in[7];
  const float* proj_b     = (const float*)d_in[8];
  const float* mg_w       = (const float*)d_in[9];
  const float* mg_b       = (const float*)d_in[10];
  const float* arb_w      = (const float*)d_in[11];
  const float* arb_b      = (const float*)d_in[12];
  const float* dec_w      = (const float*)d_in[13];
  const float* dec_b      = (const float*)d_in[14];
  const float* halt_bias  = (const float*)d_in[15];
  const float* input_gate = (const float*)d_in[16];

  float* out       = (float*)d_out;
  float* logits    = out;                                     // B*T*V
  float* stats     = out + (size_t)Bb * Tt * Vv;              // 5
  float* mem_r_out = stats + 5;                               // B*S*D
  float* mem_i_out = mem_r_out + (size_t)Bb * Ss * Dd;        // B*S*D

  const size_t FRAG_BYTES = (size_t)(Vv / 16) * 4 * 32 * 8 * 4;  // 1 MB each
  char* ws = (char*)d_ws;
  unsigned* vq_frag   = (unsigned*)ws; ws += FRAG_BYTES;
  unsigned* dec_frag  = (unsigned*)ws; ws += FRAG_BYTES;
  float*    vq_sq     = (float*)ws;    ws += (size_t)Vv * 4;
  float*    ws_flat   = (float*)ws;    ws += (size_t)Bb * Tt * K2 * 4;
  int*      idx_ws    = (int*)ws;      ws += (size_t)Tt * Rr * Bb * 4;
  int*      mask_ws   = (int*)ws;      ws += (size_t)Tt * Rr * Bb * 4;
  float*    accvq_ws  = (float*)ws;    ws += (size_t)Tt * Bb * 4;
  float*    accpond_ws= (float*)ws;    ws += (size_t)Tt * Bb * 4;
  float*    accph_ws  = (float*)ws;    ws += (size_t)Tt * Bb * 4;

  int prep_threads = Vv + 2 * ((Vv / 16) * 4 * 32);
  k_prep<<<(prep_threads + 255) / 256, 256, 0, stream>>>(vq_emb, dec_w, vq_frag, dec_frag, vq_sq);
  k_main<<<Bb / 16, 256, 0, stream>>>(x_seq, enc_emb, vq_emb, nrg, nrb, nig, nib,
                                      proj_w, proj_b, mg_w, mg_b, arb_w, arb_b,
                                      halt_bias, input_gate, vq_frag, vq_sq,
                                      ws_flat, idx_ws, mask_ws,
                                      accvq_ws, accpond_ws, accph_ws,
                                      mem_r_out, mem_i_out);
  k_decode<<<(Bb * Tt) / 16, 256, 0, stream>>>(ws_flat, dec_frag, dec_b, logits);
  k_stats<<<1, 256, 0, stream>>>(idx_ws, mask_ws, accvq_ws, accpond_ws, accph_ws, stats);
}
